// MultiHeadAttention_68951404970522
// MI455X (gfx1250) — matrix operand order repaired and checked by reading, not compiled, not run on hardware
//
#include <hip/hip_runtime.h>

#define B_  2
#define S_  2048
#define E_  1024
#define H_  16
#define D_  64
#define BS_ (B_ * S_)
#define NEG_INF_ (-1e30f)

typedef __attribute__((ext_vector_type(16))) _Float16 v16h;
typedef __attribute__((ext_vector_type(8)))  _Float16 v8h;
typedef __attribute__((ext_vector_type(8)))  float    v8f;
typedef __attribute__((ext_vector_type(4)))  float    v4f;

static __device__ __forceinline__ v16h cat8(v8h a, v8h b) {
  return __builtin_shufflevector(a, b, 0,1,2,3,4,5,6,7,8,9,10,11,12,13,14,15);
}

static __device__ __forceinline__ v8f wmma_f16(v16h a, v16h b, v8f c) {
  // D = A(16x32 f16) * B(32x16 f16) + C(16x16 f32)
  return __builtin_amdgcn_wmma_f32_16x16x32_f16(
      /*neg_a=*/false, a, /*neg_b=*/false, b,
      /*c_mod=*/(short)0, c, /*reuse_a=*/false, /*reuse_b=*/false);
}

// A-fragment (16x32, MxK) from row-major [M][K] memory, leading dim ld (elems).
// ISA layout: lane (l=lane&15, hi=lane>>4) row m=l; K = hi*8+{0..7} and 16+hi*8+{0..7}.
static __device__ __forceinline__ v16h load_a(const _Float16* base, int ld, int lane) {
  const int l = lane & 15, hi = lane >> 4;
  const _Float16* p = base + (size_t)l * ld + hi * 8;
  return cat8(*(const v8h*)p, *(const v8h*)(p + 16));
}

// B-fragment (32x16, KxN) from [N][K]-major memory (row n contiguous in K).
// ISA layout: lane column n=l; K = hi*8+{0..7} and 16+hi*8+{0..7}.
static __device__ __forceinline__ v16h load_bt(const _Float16* base, int ld, int lane) {
  const int l = lane & 15, hi = lane >> 4;
  const _Float16* p = base + (size_t)l * ld + hi * 8;
  return cat8(*(const v8h*)p, *(const v8h*)(p + 16));
}

// ------------------------------------------------ vectorized cast f32 -> f16
__global__ void cast_f16_kernel(const float* __restrict__ src,
                                _Float16* __restrict__ dst, int n8) {
  const int i = blockIdx.x * blockDim.x + threadIdx.x;
  if (i < n8) {
    const v4f x0 = *(const v4f*)(src + (size_t)i * 8);
    const v4f x1 = *(const v4f*)(src + (size_t)i * 8 + 4);
    v8h o;
#pragma unroll
    for (int j = 0; j < 4; ++j) { o[j] = (_Float16)x0[j]; o[j + 4] = (_Float16)x1[j]; }
    *(v8h*)(dst + (size_t)i * 8) = o;
  }
}

// ---------------------------------------------------------------------------
// GEMM core: 32x64 wave tile, two-stage ping-pong over K (2 x 32 per iter).
// Stage registers have fixed homes -> no accumulator evacuation copies.
// ---------------------------------------------------------------------------
#define GEMM_PING_PONG_LOOP(Ar0, Ar1, Wn0, Wn1, Wn2, Wn3, acc, lane)          \
  v16h a0 = load_a((Ar0), E_, (lane));                                        \
  v16h a1 = load_a((Ar1), E_, (lane));                                        \
  v16h b0 = load_bt((Wn0), E_, (lane));                                       \
  v16h b1 = load_bt((Wn1), E_, (lane));                                       \
  v16h b2 = load_bt((Wn2), E_, (lane));                                       \
  v16h b3 = load_bt((Wn3), E_, (lane));                                       \
  _Pragma("unroll 1")                                                         \
  for (int k0 = 0; k0 < E_; k0 += 64) {                                       \
    const int k1 = k0 + 32;                                                   \
    v16h c0 = load_a((Ar0) + k1, E_, (lane));                                 \
    v16h c1 = load_a((Ar1) + k1, E_, (lane));                                 \
    v16h d0 = load_bt((Wn0) + k1, E_, (lane));                                \
    v16h d1 = load_bt((Wn1) + k1, E_, (lane));                                \
    v16h d2 = load_bt((Wn2) + k1, E_, (lane));                                \
    v16h d3 = load_bt((Wn3) + k1, E_, (lane));                                \
    acc[0] = wmma_f16(a0, b0, acc[0]);                                        \
    acc[1] = wmma_f16(a0, b1, acc[1]);                                        \
    acc[2] = wmma_f16(a0, b2, acc[2]);                                        \
    acc[3] = wmma_f16(a0, b3, acc[3]);                                        \
    acc[4] = wmma_f16(a1, b0, acc[4]);                                        \
    acc[5] = wmma_f16(a1, b1, acc[5]);                                        \
    acc[6] = wmma_f16(a1, b2, acc[6]);                                        \
    acc[7] = wmma_f16(a1, b3, acc[7]);                                        \
    /* branchless next stage-1 load (redundant k=0 reload on last iter) */    \
    const int k2 = (k0 + 64 < E_) ? k0 + 64 : 0;                              \
    a0 = load_a((Ar0) + k2, E_, (lane));                                      \
    a1 = load_a((Ar1) + k2, E_, (lane));                                      \
    b0 = load_bt((Wn0) + k2, E_, (lane));                                     \
    b1 = load_bt((Wn1) + k2, E_, (lane));                                     \
    b2 = load_bt((Wn2) + k2, E_, (lane));                                     \
    b3 = load_bt((Wn3) + k2, E_, (lane));                                     \
    acc[0] = wmma_f16(c0, d0, acc[0]);                                        \
    acc[1] = wmma_f16(c0, d1, acc[1]);                                        \
    acc[2] = wmma_f16(c0, d2, acc[2]);                                        \
    acc[3] = wmma_f16(c0, d3, acc[3]);                                        \
    acc[4] = wmma_f16(c1, d0, acc[4]);                                        \
    acc[5] = wmma_f16(c1, d1, acc[5]);                                        \
    acc[6] = wmma_f16(c1, d2, acc[6]);                                        \
    acc[7] = wmma_f16(c1, d3, acc[7]);                                        \
  }

// ------------------------------------------------- fused Q/K/V projection GEMM
// y[m][n] = sum_k xh[m][k] * W[n][k];  z = blockIdx.z selects Wq/Wk/Wv.
// Q,K stored [b,h,s,d]; V stored transposed [b,h,d,s].
__global__ __launch_bounds__(256) void proj_qkv_kernel(
    const _Float16* __restrict__ xh,
    const _Float16* __restrict__ wqh, const _Float16* __restrict__ wkh,
    const _Float16* __restrict__ wvh,
    _Float16* __restrict__ qh, _Float16* __restrict__ kh,
    _Float16* __restrict__ vth) {
  const int lane = threadIdx.x & 31;
  const int wid  = threadIdx.x >> 5;
  const int m0   = blockIdx.x * 256 + wid * 32;
  const int n0   = blockIdx.y * 64;
  const int z    = blockIdx.z;
  const _Float16* W = (z == 0) ? wqh : (z == 1) ? wkh : wvh;

  const _Float16* Ar0 = xh + (size_t)m0 * E_;
  const _Float16* Ar1 = xh + (size_t)(m0 + 16) * E_;
  const _Float16* Wn0 = W + (size_t)(n0 +  0) * E_;
  const _Float16* Wn1 = W + (size_t)(n0 + 16) * E_;
  const _Float16* Wn2 = W + (size_t)(n0 + 32) * E_;
  const _Float16* Wn3 = W + (size_t)(n0 + 48) * E_;

  v8f acc[8] = {};
  GEMM_PING_PONG_LOOP(Ar0, Ar1, Wn0, Wn1, Wn2, Wn3, acc, lane)

  const int l = lane & 15, hi = lane >> 4;
#pragma unroll
  for (int mt = 0; mt < 2; ++mt) {
#pragma unroll
    for (int nt = 0; nt < 4; ++nt) {
      const v8f a = acc[mt * 4 + nt];
#pragma unroll
      for (int r = 0; r < 8; ++r) {
        const int m = m0 + mt * 16 + r + 8 * hi;   // token row
        const int n = n0 + nt * 16 + l;            // embed col
        const int b = m >> 11;                     // m / S_
        const int s = m & (S_ - 1);
        const int h = n >> 6;                      // n / D_
        const int d = n & (D_ - 1);
        const size_t bh = (size_t)(b * H_ + h);
        const _Float16 v = (_Float16)a[r];
        if (z == 0)      qh [(bh * S_ + s) * D_ + d] = v;
        else if (z == 1) kh [(bh * S_ + s) * D_ + d] = v;
        else             vth[(bh * D_ + d) * S_ + s] = v;
      }
    }
  }
}

// --------------------------------------------------- flash attention (per wave)
// One wave: 16-row q tile, online softmax over 32-wide causal k blocks.
// K fragments for block kb+1 are prefetched before the softmax of block kb.
__global__ __launch_bounds__(128) void attn_kernel(
    const _Float16* __restrict__ qh, const _Float16* __restrict__ kh,
    const _Float16* __restrict__ vth, _Float16* __restrict__ oh) {
  __shared__ float    sS[4][16 * 33];   // f32 scores, padded stride
  __shared__ _Float16 sP[4][16 * 32];   // f16 probabilities, A-frag source
  __shared__ float    sM[4][16], sL[4][16], sC[4][16];

  const int lane = threadIdx.x & 31;
  const int wid  = threadIdx.x >> 5;
  const int tile = blockIdx.x * 4 + wid;      // 0..4095
  const int qt   = tile & (S_ / 16 - 1);      // q tile within sequence
  const int bh   = tile >> 7;                 // b*H + h
  const int h    = bh & (H_ - 1);
  const int q0   = qt * 16;
  const float slope = exp2f(-0.5f * (float)(h + 1));   // 2^(-8*(h+1)/16)

  float*    Ss = sS[wid];
  _Float16* Ps = sP[wid];
  float *Ms = sM[wid], *Ls = sL[wid], *Cs = sC[wid];

  const int l = lane & 15, hi = lane >> 4;

  // Q A-fragments for d = 0..31 and 32..63 (held for whole loop)
  const _Float16* Qb = qh + ((size_t)bh * S_ + q0) * D_;
  const v16h qa0 = load_a(Qb + 0,  D_, lane);
  const v16h qa1 = load_a(Qb + 32, D_, lane);

  const _Float16* Kh = kh + (size_t)bh * S_ * D_;   // [s][d] for this head
  const _Float16* Vt = vth + (size_t)bh * D_ * S_;  // [d][s] for this head

  // prologue: K fragments for kb = 0
  v16h kf0 = load_bt(Kh + 0 * D_ + 0,  D_, lane);
  v16h kf1 = load_bt(Kh + 0 * D_ + 32, D_, lane);
  v16h kf2 = load_bt(Kh + 16 * D_ + 0,  D_, lane);
  v16h kf3 = load_bt(Kh + 16 * D_ + 32, D_, lane);

  v8f o0 = {}, o1 = {}, o2 = {}, o3 = {};
  if (lane < 16) { Ms[lane] = NEG_INF_; Ls[lane] = 0.0f; }
  asm volatile("s_wait_dscnt 0x0" ::: "memory");

  const int kbmax = (q0 + 15) >> 5;           // inclusive causal block limit
#pragma unroll 1
  for (int kb = 0; kb <= kbmax; ++kb) {
    const int ks = kb * 32;

    // -------- scores: two 16x16 tiles from prefetched K fragments
    v8f sc0 = {}, sc1 = {};
    sc0 = wmma_f16(qa0, kf0, sc0);
    sc0 = wmma_f16(qa1, kf1, sc0);
    sc1 = wmma_f16(qa0, kf2, sc1);
    sc1 = wmma_f16(qa1, kf3, sc1);

    // -------- issue V loads (this block) + K loads (next block) NOW:
    // their latency hides behind the softmax below.
    const v16h vb0 = load_bt(Vt +  0 * S_ + ks, S_, lane);
    const v16h vb1 = load_bt(Vt + 16 * S_ + ks, S_, lane);
    const v16h vb2 = load_bt(Vt + 32 * S_ + ks, S_, lane);
    const v16h vb3 = load_bt(Vt + 48 * S_ + ks, S_, lane);
    const int ksn = (kb < kbmax) ? ks + 32 : ks;   // branchless clamp
    kf0 = load_bt(Kh + (size_t)(ksn +  0) * D_ + 0,  D_, lane);
    kf1 = load_bt(Kh + (size_t)(ksn +  0) * D_ + 32, D_, lane);
    kf2 = load_bt(Kh + (size_t)(ksn + 16) * D_ + 0,  D_, lane);
    kf3 = load_bt(Kh + (size_t)(ksn + 16) * D_ + 32, D_, lane);

    // -------- scale + ALiBi + causal mask, scores -> LDS
#pragma unroll
    for (int r = 0; r < 8; ++r) {
      const int qrow = q0 + r + 8 * hi;
      const int kc0  = ks + l;
      const int kc1  = ks + 16 + l;
      float v0 = sc0[r] * 0.125f + slope * (float)(kc0 - qrow);
      float v1 = sc1[r] * 0.125f + slope * (float)(kc1 - qrow);
      if (kc0 > qrow) v0 = NEG_INF_;
      if (kc1 > qrow) v1 = NEG_INF_;
      Ss[(r + 8 * hi) * 33 +      l] = v0;
      Ss[(r + 8 * hi) * 33 + 16 + l] = v1;
    }
    asm volatile("s_wait_dscnt 0x0" ::: "memory");

    // -------- online softmax row update (lanes 0..15 own one row each)
    if (lane < 16) {
      const int row = lane;
      const float mold = Ms[row];
      float mnew = mold;
      float sv[32];
#pragma unroll
      for (int j = 0; j < 32; ++j) { sv[j] = Ss[row * 33 + j]; mnew = fmaxf(mnew, sv[j]); }
      const float corr = __expf(mold - mnew);
      float sum = 0.0f;
#pragma unroll
      for (int j = 0; j < 32; ++j) {
        const float p = __expf(sv[j] - mnew);
        sum += p;
        Ps[row * 32 + j] = (_Float16)p;
      }
      Ls[row] = Ls[row] * corr + sum;
      Ms[row] = mnew;
      Cs[row] = corr;
    }
    asm volatile("s_wait_dscnt 0x0" ::: "memory");

    // -------- rescale O accumulators by row correction
    float c[8];
#pragma unroll
    for (int r = 0; r < 8; ++r) c[r] = Cs[8 * hi + r];
#pragma unroll
    for (int r = 0; r < 8; ++r) {
      o0[r] *= c[r]; o1[r] *= c[r]; o2[r] *= c[r]; o3[r] *= c[r];
    }

    // -------- P as A-fragment from LDS (row l, K chunks hi*8 and 16+hi*8)
    const _Float16* pp = Ps + l * 32 + hi * 8;
    const v16h pa = cat8(*(const v8h*)pp, *(const v8h*)(pp + 16));

    // -------- O += P @ V  (V fragments already resident)
    o0 = wmma_f16(pa, vb0, o0);
    o1 = wmma_f16(pa, vb1, o1);
    o2 = wmma_f16(pa, vb2, o2);
    o3 = wmma_f16(pa, vb3, o3);
  }

  // -------- finalize: divide by row sums, store O f16 as [token][E]
  float invl[8];
#pragma unroll
  for (int r = 0; r < 8; ++r) invl[r] = 1.0f / Ls[8 * hi + r];
  const size_t tok0 = (size_t)(bh >> 4) * S_ + q0;
#pragma unroll
  for (int r = 0; r < 8; ++r) {
    _Float16* orow = oh + (tok0 + r + 8 * hi) * E_ + h * D_;
    orow[ 0 + l] = (_Float16)(o0[r] * invl[r]);
    orow[16 + l] = (_Float16)(o1[r] * invl[r]);
    orow[32 + l] = (_Float16)(o2[r] * invl[r]);
    orow[48 + l] = (_Float16)(o3[r] * invl[r]);
  }
}

// ------------------------------------------------------ output GEMM: O@Wo.T+bo
__global__ __launch_bounds__(256) void out_gemm_kernel(
    const _Float16* __restrict__ oh, const _Float16* __restrict__ woh,
    const float* __restrict__ bo, float* __restrict__ out) {
  const int lane = threadIdx.x & 31;
  const int wid  = threadIdx.x >> 5;
  const int m0   = blockIdx.x * 256 + wid * 32;
  const int n0   = blockIdx.y * 64;

  const _Float16* Ar0 = oh + (size_t)m0 * E_;
  const _Float16* Ar1 = oh + (size_t)(m0 + 16) * E_;
  const _Float16* Wn0 = woh + (size_t)(n0 +  0) * E_;
  const _Float16* Wn1 = woh + (size_t)(n0 + 16) * E_;
  const _Float16* Wn2 = woh + (size_t)(n0 + 32) * E_;
  const _Float16* Wn3 = woh + (size_t)(n0 + 48) * E_;

  v8f acc[8] = {};
  GEMM_PING_PONG_LOOP(Ar0, Ar1, Wn0, Wn1, Wn2, Wn3, acc, lane)

  const int l = lane & 15, hi = lane >> 4;
#pragma unroll
  for (int mt = 0; mt < 2; ++mt) {
#pragma unroll
    for (int nt = 0; nt < 4; ++nt) {
      const v8f a = acc[mt * 4 + nt];
      const int n = n0 + nt * 16 + l;
      const float bias = bo[n];
#pragma unroll
      for (int r = 0; r < 8; ++r) {
        out[(size_t)(m0 + mt * 16 + r + 8 * hi) * E_ + n] = a[r] + bias;
      }
    }
  }
}

// ------------------------------------------------------------------- launcher
extern "C" void kernel_launch(void* const* d_in, const int* in_sizes, int n_in,
                              void* d_out, int out_size, void* d_ws, size_t ws_size,
                              hipStream_t stream) {
  (void)in_sizes; (void)n_in; (void)out_size; (void)ws_size;
  const float* x  = (const float*)d_in[0];
  const float* Wq = (const float*)d_in[1];
  const float* Wk = (const float*)d_in[2];
  const float* Wv = (const float*)d_in[3];
  const float* Wo = (const float*)d_in[4];
  const float* bo = (const float*)d_in[5];
  float* out = (float*)d_out;

  // workspace layout (bytes)
  char* ws = (char*)d_ws;
  const size_t XH  = (size_t)BS_ * E_ * sizeof(_Float16);          // 8 MiB
  const size_t WH  = (size_t)E_ * E_ * sizeof(_Float16);           // 2 MiB
  const size_t QKV = (size_t)B_ * H_ * S_ * D_ * sizeof(_Float16); // 8 MiB
  _Float16* xh  = (_Float16*)(ws);
  _Float16* wqh = (_Float16*)(ws + XH);
  _Float16* wkh = (_Float16*)(ws + XH + WH);
  _Float16* wvh = (_Float16*)(ws + XH + 2 * WH);
  _Float16* woh = (_Float16*)(ws + XH + 3 * WH);
  _Float16* qh  = (_Float16*)(ws + XH + 4 * WH);
  _Float16* kh  = (_Float16*)(ws + XH + 4 * WH + QKV);
  _Float16* vth = (_Float16*)(ws + XH + 4 * WH + 2 * QKV);
  _Float16* oh  = (_Float16*)(ws + XH + 4 * WH + 3 * QKV);

  const int nx8 = (BS_ * E_) / 8;   // 524,288
  const int nw8 = (E_ * E_) / 8;    // 131,072
  cast_f16_kernel<<<(nx8 + 255) / 256, 256, 0, stream>>>(x,  xh,  nx8);
  cast_f16_kernel<<<(nw8 + 255) / 256, 256, 0, stream>>>(Wq, wqh, nw8);
  cast_f16_kernel<<<(nw8 + 255) / 256, 256, 0, stream>>>(Wk, wkh, nw8);
  cast_f16_kernel<<<(nw8 + 255) / 256, 256, 0, stream>>>(Wv, wvh, nw8);
  cast_f16_kernel<<<(nw8 + 255) / 256, 256, 0, stream>>>(Wo, woh, nw8);

  proj_qkv_kernel<<<dim3(BS_ / 256, E_ / 64, 3), 256, 0, stream>>>(
      xh, wqh, wkh, wvh, qh, kh, vth);

  // B*H*(S/16) = 4096 wave-tiles, 4 waves per block
  attn_kernel<<<(B_ * H_ * (S_ / 16)) / 4, 128, 0, stream>>>(qh, kh, vth, oh);

  out_gemm_kernel<<<dim3(BS_ / 256, E_ / 64), 256, 0, stream>>>(oh, woh, bo, out);
}
